// SemanticLevelContext_66468913873215
// MI455X (gfx1250) — matrix operand order repaired and checked by reading, not compiled, not run on hardware
//
#include <hip/hip_runtime.h>
#include <hip/hip_bf16.h>
#include <math.h>

// Problem constants (from reference): B=2, C=64, D=12, H=20, W=40, CT=32
#define B_    2
#define C_    64
#define D_    12
#define HW_   800          // H*W
#define NPB_  9600         // D*H*W, tokens per batch
#define NTOT_ 19200        // B * NPB_
#define CT_   32
#define NQT_  600          // query tiles of 16 rows per batch

typedef __attribute__((ext_vector_type(16))) _Float16 v16h;
typedef __attribute__((ext_vector_type(8)))  _Float16 v8h;
typedef __attribute__((ext_vector_type(8)))  float    v8f;

__device__ __forceinline__ v16h make16(v8h lo, v8h hi) {
  v16h r;
#pragma unroll
  for (int i = 0; i < 8; ++i) { r[i] = lo[i]; r[8 + i] = hi[i]; }
  return r;
}

// DPP16 row_xmask lane swap within 16-lane rows: ctrl = 0x160 | xor_mask.
// Pure VALU (v_mov_b32_dpp) -- replaces ds_bpermute butterflies.
template <int CTRL>
__device__ __forceinline__ float dpp_mov(float v) {
  return __int_as_float(
      __builtin_amdgcn_update_dpp(0, __float_as_int(v), CTRL, 0xF, 0xF, true));
}
__device__ __forceinline__ float redmax16(float v) {
  v = fmaxf(v, dpp_mov<0x161>(v));
  v = fmaxf(v, dpp_mov<0x162>(v));
  v = fmaxf(v, dpp_mov<0x164>(v));
  v = fmaxf(v, dpp_mov<0x168>(v));
  return v;
}
__device__ __forceinline__ float redsum16(float v) {
  v += dpp_mov<0x161>(v);
  v += dpp_mov<0x162>(v);
  v += dpp_mov<0x164>(v);
  v += dpp_mov<0x168>(v);
  return v;
}

// ---------------------------------------------------------------------------
// 1) Per-pixel softmax over D + argmax + gathered prob
// ---------------------------------------------------------------------------
__global__ void k_softargmax(const float* __restrict__ preds,
                             int* __restrict__ aArr, float* __restrict__ pvArr) {
  int idx = blockIdx.x * blockDim.x + threadIdx.x;
  if (idx >= B_ * HW_) return;
  int b = idx / HW_, pix = idx % HW_;
  const float* pr = preds + (size_t)b * D_ * HW_ + pix;
  float mx = -1e30f; int am = 0;
#pragma unroll
  for (int d = 0; d < D_; ++d) {
    float v = pr[(size_t)d * HW_];
    if (v > mx) { mx = v; am = d; }
  }
  float s = 0.f;
#pragma unroll
  for (int d = 0; d < D_; ++d) s += __expf(pr[(size_t)d * HW_] - mx);
  aArr[idx] = am;
  pvArr[idx] = 1.0f / s;  // exp(pr[am]-mx)==1
}

// ---------------------------------------------------------------------------
// 2) Segment max / sum(exp) per (batch, disparity class) via LDS atomics
// ---------------------------------------------------------------------------
__global__ void k_segstats(const int* __restrict__ aArr, const float* __restrict__ pvArr,
                           float* __restrict__ segmax, float* __restrict__ segsum) {
  __shared__ int   smax[D_];
  __shared__ float ssum[D_];
  int b = blockIdx.x, tid = threadIdx.x;
  if (tid < D_) { smax[tid] = __float_as_int(-1e30f); ssum[tid] = 0.f; }
  __syncthreads();
  for (int pix = tid; pix < HW_; pix += blockDim.x) {
    int a = aArr[b * HW_ + pix];
    atomicMax(&smax[a], __float_as_int(pvArr[b * HW_ + pix]));  // pv > 0 -> int order ok
  }
  __syncthreads();
  for (int pix = tid; pix < HW_; pix += blockDim.x) {
    int a = aArr[b * HW_ + pix];
    atomicAdd(&ssum[a], __expf(pvArr[b * HW_ + pix] - __int_as_float(smax[a])));
  }
  __syncthreads();
  if (tid < D_) {
    segmax[b * D_ + tid] = __int_as_float(smax[tid]);
    segsum[b * D_ + tid] = ssum[tid];
  }
}

// ---------------------------------------------------------------------------
// 3) key_feats = x + scatter(argmax-plane * segment-softmax weight)
//    written channel-major: keyf[c][b*NPB + d*HW + pix]
// ---------------------------------------------------------------------------
__global__ void k_keyfeats(const float* __restrict__ x, const int* __restrict__ aArr,
                           const float* __restrict__ pvArr, const float* __restrict__ segmax,
                           const float* __restrict__ segsum, float* __restrict__ keyf) {
  int idx = blockIdx.x * blockDim.x + threadIdx.x;
  if (idx >= B_ * C_ * D_ * HW_) return;
  int pix = idx % HW_;
  int t = idx / HW_;
  int dd = t % D_;  t /= D_;
  int c = t % C_;
  int b = t / C_;
  int a = aArr[b * HW_ + pix];
  float xv = x[(((size_t)b * C_ + c) * D_ + dd) * HW_ + pix];
  float add = 0.f;
  if (dd == a) {
    float wt = __expf(pvArr[b * HW_ + pix] - segmax[b * D_ + a]) / segsum[b * D_ + a];
    add = xv * wt;  // x at argmax plane == xv here
  }
  int pos = b * NPB_ + dd * HW_ + pix;
  keyf[(size_t)c * NTOT_ + pos] = xv + add;
}

// ---------------------------------------------------------------------------
// 4) conv_bn_relu pass 1: Y = W(OxC) @ X(CxN) + per-channel sum/sumsq atomics
//    inMode 0: X[c*NTOT + pos]   inMode 1: X[(b*C + c)*NPB + p]  (raw x layout)
// ---------------------------------------------------------------------------
__global__ void k_gemm_stats(const float* __restrict__ W, const float* __restrict__ X,
                             float* __restrict__ Y, float* __restrict__ stats,
                             int O, int Cc, int inMode) {
  __shared__ float s1[256], s2[256];
  int tid = threadIdx.x;
  int pos = blockIdx.x * 256 + tid;
  int o = blockIdx.y;
  float acc = 0.f;
  if (pos < NTOT_) {
    if (inMode == 0) {
      const float* xp = X + pos;
      for (int c = 0; c < Cc; ++c) acc = fmaf(W[o * Cc + c], xp[(size_t)c * NTOT_], acc);
    } else {
      int b = pos / NPB_, p = pos % NPB_;
      const float* xp = X + (size_t)b * Cc * NPB_ + p;
      for (int c = 0; c < Cc; ++c) acc = fmaf(W[o * Cc + c], xp[(size_t)c * NPB_], acc);
    }
    Y[(size_t)o * NTOT_ + pos] = acc;
  }
  s1[tid] = (pos < NTOT_) ? acc : 0.f;
  s2[tid] = (pos < NTOT_) ? acc * acc : 0.f;
  __syncthreads();
  for (int st = 128; st > 0; st >>= 1) {
    if (tid < st) { s1[tid] += s1[tid + st]; s2[tid] += s2[tid + st]; }
    __syncthreads();
  }
  if (tid == 0) { atomicAdd(&stats[o], s1[0]); atomicAdd(&stats[O + o], s2[0]); }
}

// ---------------------------------------------------------------------------
// 5) conv_bn_relu pass 2: normalize + ReLU, emit in attention-friendly layout
//    mode 0: f32 [o*NTOT+pos]            mode 1: f16 n-major [pos*O + o]   (q,k)
//    mode 2: f16 c-major [o*NTOT+pos](v) mode 3: f32 [(b*O+o)*NPB + p] (final out)
//    preScale folds the attention 1/sqrt(CT) into the q conversion.
// ---------------------------------------------------------------------------
__global__ void k_bnrelu(const float* __restrict__ Y, const float* __restrict__ stats,
                         int O, int mode, float preScale, float* __restrict__ fout,
                         _Float16* __restrict__ hout) {
  int idx = blockIdx.x * 256 + threadIdx.x;
  if (idx >= O * NTOT_) return;
  int o = idx / NTOT_, pos = idx % NTOT_;
  float inv_n = 1.f / (float)NTOT_;
  float mu = stats[o] * inv_n;
  float var = stats[O + o] * inv_n - mu * mu;
  float rs = rsqrtf(var + 1e-5f);
  float v = (Y[idx] - mu) * rs;
  v = v > 0.f ? v : 0.f;
  if (mode == 0)      fout[idx] = v;
  else if (mode == 1) hout[(size_t)pos * O + o] = (_Float16)(v * preScale);
  else if (mode == 2) hout[idx] = (_Float16)(v * preScale);
  else { int b = pos / NPB_, p = pos % NPB_; fout[((size_t)b * O + o) * NPB_ + p] = v; }
}

// ---------------------------------------------------------------------------
// 6) Flash attention: one wave per 16-query tile, 64 keys per chunk.
//    CT=32 == WMMA K-dim. q (pre-scaled),k: n-major f16; vT: ct-major f16.
//    Softmax row reductions use DPP row_xmask (VALU), not LDS bpermute.
//    ctx out: f32 channel-major (CT, NTOT).
// ---------------------------------------------------------------------------
__global__ __launch_bounds__(32) void k_flash(const _Float16* __restrict__ q,
                                              const _Float16* __restrict__ k,
                                              const _Float16* __restrict__ vT,
                                              float* __restrict__ ctx) {
  __shared__ __align__(32) _Float16 pl[16 * 64];
  const int lane = threadIdx.x;
  const int half = lane >> 4;   // which K-half this lane packs
  const int l15  = lane & 15;
  const int b  = blockIdx.x / NQT_;
  const int qt = blockIdx.x % NQT_;
  const int q0 = qt * 16;

  // Q tile in WMMA A layout: lane holds row l15; elems[0..15]=K[half*16..half*16+15]
  // (q already scaled by 1/sqrt(CT))
  const _Float16* qr = q + (size_t)(b * NPB_ + q0 + l15) * CT_;
  v16h aq = make16(*(const v8h*)(qr + half * 16), *(const v8h*)(qr + half * 16 + 8));

  v8f o0 = {}; v8f o1 = {};
  float m[8], l[8];
#pragma unroll
  for (int r = 0; r < 8; ++r) { m[r] = -1e30f; l[r] = 0.f; }

  for (int kb = 0; kb < NPB_; kb += 64) {
    // K^T B-tiles (4 x 16 keys): lane = key column; lanes 0-15 carry ct 0..15,
    // lanes 16-31 carry ct 16..31
    const _Float16* kp = k + (size_t)(b * NPB_ + kb + l15) * CT_ + half * 16;
    v16h bk0 = make16(*(const v8h*)(kp),                *(const v8h*)(kp + 8));
    v16h bk1 = make16(*(const v8h*)(kp + 16 * CT_),     *(const v8h*)(kp + 16 * CT_ + 8));
    v16h bk2 = make16(*(const v8h*)(kp + 32 * CT_),     *(const v8h*)(kp + 32 * CT_ + 8));
    v16h bk3 = make16(*(const v8h*)(kp + 48 * CT_),     *(const v8h*)(kp + 48 * CT_ + 8));
    v8f z = {};
    v8f s0 = __builtin_amdgcn_wmma_f32_16x16x32_f16(false, aq, false, bk0, (short)0, z, false, false);
    v8f s1 = __builtin_amdgcn_wmma_f32_16x16x32_f16(false, aq, false, bk1, (short)0, z, false, false);
    v8f s2 = __builtin_amdgcn_wmma_f32_16x16x32_f16(false, aq, false, bk2, (short)0, z, false, false);
    v8f s3 = __builtin_amdgcn_wmma_f32_16x16x32_f16(false, aq, false, bk3, (short)0, z, false, false);

    // online softmax: D-layout spreads each row over a 16-lane half
#pragma unroll
    for (int r = 0; r < 8; ++r) {
      float a0 = s0[r], a1 = s1[r], a2 = s2[r], a3 = s3[r];
      float rm = redmax16(fmaxf(fmaxf(a0, a1), fmaxf(a2, a3)));
      float mn = fmaxf(m[r], rm);
      float alpha = __expf(m[r] - mn);
      float p0 = __expf(a0 - mn), p1 = __expf(a1 - mn);
      float p2 = __expf(a2 - mn), p3 = __expf(a3 - mn);
      float rsum = redsum16((p0 + p1) + (p2 + p3));
      l[r] = l[r] * alpha + rsum;
      m[r] = mn;
      o0[r] *= alpha; o1[r] *= alpha;
      _Float16* prow = pl + (r + half * 8) * 64 + l15;
      prow[0]  = (_Float16)p0;
      prow[16] = (_Float16)p1;
      prow[32] = (_Float16)p2;
      prow[48] = (_Float16)p3;
    }
    __syncthreads();   // 1-wave workgroup: lowers to a cheap wait, not a real barrier
    // reload P in A layout: ap0 = keys 0..31 of chunk, ap1 = keys 32..63
    const _Float16* prd = pl + l15 * 64;
    v16h ap0 = make16(*(const v8h*)(prd + half * 16),      *(const v8h*)(prd + 8 + half * 16));
    v16h ap1 = make16(*(const v8h*)(prd + 32 + half * 16), *(const v8h*)(prd + 40 + half * 16));
    __syncthreads();

    // V B-tiles: lane = ct column; K-dim = 32 keys (contiguous in vT row)
    const _Float16* v0p = vT + (size_t)l15 * NTOT_ + b * NPB_ + kb + half * 16;
    const _Float16* v1p = vT + (size_t)(16 + l15) * NTOT_ + b * NPB_ + kb + half * 16;
    v16h bv00 = make16(*(const v8h*)v0p,        *(const v8h*)(v0p + 8));
    v16h bv01 = make16(*(const v8h*)v1p,        *(const v8h*)(v1p + 8));
    v16h bv10 = make16(*(const v8h*)(v0p + 32), *(const v8h*)(v0p + 40));
    v16h bv11 = make16(*(const v8h*)(v1p + 32), *(const v8h*)(v1p + 40));
    o0 = __builtin_amdgcn_wmma_f32_16x16x32_f16(false, ap0, false, bv00, (short)0, o0, false, false);
    o0 = __builtin_amdgcn_wmma_f32_16x16x32_f16(false, ap1, false, bv10, (short)0, o0, false, false);
    o1 = __builtin_amdgcn_wmma_f32_16x16x32_f16(false, ap0, false, bv01, (short)0, o1, false, false);
    o1 = __builtin_amdgcn_wmma_f32_16x16x32_f16(false, ap1, false, bv11, (short)0, o1, false, false);
  }

#pragma unroll
  for (int r = 0; r < 8; ++r) {
    float inv = 1.f / l[r];
    int pos = b * NPB_ + q0 + r + half * 8;
    ctx[(size_t)l15 * NTOT_ + pos]        = o0[r] * inv;
    ctx[(size_t)(16 + l15) * NTOT_ + pos] = o1[r] * inv;
  }
}

// ---------------------------------------------------------------------------
// Host launcher
// ---------------------------------------------------------------------------
extern "C" void kernel_launch(void* const* d_in, const int* in_sizes, int n_in,
                              void* d_out, int out_size, void* d_ws, size_t ws_size,
                              hipStream_t stream) {
  (void)in_sizes; (void)n_in; (void)out_size; (void)ws_size;
  const float* x     = (const float*)d_in[0];
  const float* preds = (const float*)d_in[1];
  const float* Wq1   = (const float*)d_in[2];
  const float* Wq2   = (const float*)d_in[3];
  const float* Wk1   = (const float*)d_in[4];
  const float* Wk2   = (const float*)d_in[5];
  const float* Wv    = (const float*)d_in[6];
  const float* Wo    = (const float*)d_in[7];
  float* out = (float*)d_out;

  char* ws = (char*)d_ws;
  size_t off = 0;
  auto carve = [&](size_t bytes) -> void* {
    void* p = ws + off;
    off += (bytes + 255) & ~(size_t)255;
    return p;
  };
  int*       aArr   = (int*)carve((size_t)B_ * HW_ * 4);
  float*     pvArr  = (float*)carve((size_t)B_ * HW_ * 4);
  float*     segmax = (float*)carve((size_t)B_ * D_ * 4);
  float*     segsum = (float*)carve((size_t)B_ * D_ * 4);
  float*     keyf   = (float*)carve((size_t)C_ * NTOT_ * 4);
  float*     bufP   = (float*)carve((size_t)C_ * NTOT_ * 4);   // gemm Y (max O=64)
  float*     bufQ   = (float*)carve((size_t)CT_ * NTOT_ * 4);  // normalized f32 intermediate
  _Float16*  qh     = (_Float16*)carve((size_t)NTOT_ * CT_ * 2);
  _Float16*  kh     = (_Float16*)carve((size_t)NTOT_ * CT_ * 2);
  _Float16*  vh     = (_Float16*)carve((size_t)NTOT_ * CT_ * 2);
  float*     ctx    = (float*)carve((size_t)CT_ * NTOT_ * 4);
  float*     stats  = (float*)carve((size_t)2 * C_ * 4);

  // semantic gather
  k_softargmax<<<(B_ * HW_ + 255) / 256, 256, 0, stream>>>(preds, aArr, pvArr);
  k_segstats<<<B_, 256, 0, stream>>>(aArr, pvArr, segmax, segsum);
  {
    int tot = B_ * C_ * D_ * HW_;
    k_keyfeats<<<(tot + 255) / 256, 256, 0, stream>>>(x, aArr, pvArr, segmax, segsum, keyf);
  }

  auto cbr = [&](const float* Wm, const float* Xin, int O, int Cc, int inMode,
                 int outMode, float preScale, float* fout, _Float16* hout) {
    hipMemsetAsync(stats, 0, 2 * C_ * sizeof(float), stream);
    dim3 g((NTOT_ + 255) / 256, O);
    k_gemm_stats<<<g, 256, 0, stream>>>(Wm, Xin, bufP, stats, O, Cc, inMode);
    int tot = O * NTOT_;
    k_bnrelu<<<(tot + 255) / 256, 256, 0, stream>>>(bufP, stats, O, outMode, preScale, fout, hout);
  };

  const float qscale = 0.17677669529663687f;  // 1/sqrt(CT)

  // q = cbr(cbr(x, Wq1), Wq2)           -> f16 n-major, pre-scaled by 1/sqrt(CT)
  cbr(Wq1, x,    CT_, C_,  1, 0, 1.0f,   bufQ, nullptr);
  cbr(Wq2, bufQ, CT_, CT_, 0, 1, qscale, nullptr, qh);
  // k = cbr(cbr(key_feats, Wk1), Wk2)   -> f16 n-major
  cbr(Wk1, keyf, CT_, C_,  0, 0, 1.0f,   bufQ, nullptr);
  cbr(Wk2, bufQ, CT_, CT_, 0, 1, 1.0f,   nullptr, kh);
  // v = cbr(key_feats, Wv)              -> f16 ct-major (vT)
  cbr(Wv,  keyf, CT_, C_,  0, 2, 1.0f,   nullptr, vh);

  // flash attention: sim softmax + ctx, never materializing the 9600x9600 scores
  k_flash<<<B_ * NQT_, 32, 0, stream>>>(qh, kh, vh, ctx);

  // out = cbr(ctx, Wo) -> (b, 64, d, h, w) f32
  cbr(Wo, ctx, C_, CT_, 0, 3, 1.0f, out, nullptr);
}
